// GuidedPAKA2d_53944789238140
// MI455X (gfx1250) — compile-verified, matrix operand order checked by reading, not run on hardware
//
#include <hip/hip_runtime.h>

typedef _Float16 v2h  __attribute__((ext_vector_type(2)));
typedef _Float16 v8h  __attribute__((ext_vector_type(8)));
typedef _Float16 v16h __attribute__((ext_vector_type(16)));
typedef float    v4f  __attribute__((ext_vector_type(4)));
typedef float    v8f  __attribute__((ext_vector_type(8)));

namespace {
constexpr int HW     = 128;        // image h == w
constexpr int C      = 64;         // input channels
constexpr int OC     = 64;         // output channels
constexpr int K9     = 9;          // 3x3 taps
constexpr int KTOT   = C * K9;     // 576; K ordering: Kg = k9*64 + c
constexpr int NSTEP  = KTOT / 32;  // 18 WMMA K-steps
constexpr int NFRAG  = 4 * NSTEP * 32 * 16;       // halves in fragment buffer
constexpr size_t WFRAG_BYTES = (size_t)NFRAG * 2; // 73728
constexpr int OSTR   = HW + 4;     // padded lds_out row stride (dwords)
constexpr float NL2E = -2.8853900817779268f;      // -2*log2(e)
}

// weight fragment element (pre-swizzled for WMMA B operand)
// idx = ((og*NSTEP + s)*32 + lane)*16 + j ; holds weight[o = og*16+lane%16][Kg]
// Kg = s*32 + koff, koff = j + 8*((j>>3) + lane/16); c = Kg&63, k9 = Kg>>6
__device__ __forceinline__ _Float16 wfrag_elem(const float* __restrict__ wt, int idx) {
    int j    = idx & 15;
    int ln   = (idx >> 4) & 31;
    int s    = (idx >> 9) % NSTEP;
    int og   = idx / (512 * NSTEP);
    int half = ln >> 4;
    int o    = og * 16 + (ln & 15);
    int koff = j + 8 * ((j >> 3) + half);
    int kg   = s * 32 + koff;
    int c    = kg & (C - 1);
    int k9   = kg >> 6;
    return (_Float16)wt[(o * C + c) * K9 + k9];
}

__global__ __launch_bounds__(256)
void paka_prep_kernel(const float* __restrict__ wt, _Float16* __restrict__ wfg) {
    int idx = blockIdx.x * 256 + threadIdx.x;
    if (idx < NFRAG) wfg[idx] = wfrag_elem(wt, idx);
}

template <bool FROM_WS>
__global__ __launch_bounds__(256)
void paka_wmma_kernel(const float* __restrict__ x,
                      const float* __restrict__ gc,
                      const float* __restrict__ gs,
                      const float* __restrict__ wt,
                      const _Float16* __restrict__ wfg,
                      float* __restrict__ out)
{
    __shared__ _Float16 wfrag[NFRAG];      // 72 KB weight fragments
    __shared__ float    lds_out[OC * OSTR]; // 33.8 KB store-transpose buffer

    const int t    = threadIdx.x;
    const int lane = t & 31;
    const int wid  = t >> 5;             // 8 waves; wave owns 16 pixels x 64 outs
    const int b    = blockIdx.x >> 7;    // one (b,h) image row per block
    const int h    = blockIdx.x & (HW - 1);

    if (FROM_WS) {
        // coalesced 72 KB copy of pre-swizzled fragments
        const v8h* src = (const v8h*)wfg;
        v8h*       dst = (v8h*)wfrag;
        #pragma unroll
        for (int i = 0; i < NFRAG / 8 / 256; ++i)
            dst[i * 256 + t] = src[i * 256 + t];
    } else {
        for (int idx = t; idx < NFRAG; idx += 256)
            wfrag[idx] = wfrag_elem(wt, idx);
    }
    __syncthreads();

    const int m    = lane & 15;
    const int half = lane >> 4;
    const int pix  = wid * 16 + m;       // this lane's pixel (A-matrix row)

    // per-lane channel-guide cache: c(ks,q,i) = ks*32 + q*16 + 8*half + i
    // gq = -2log2e*gc - 1 (prebiased exp2 argument)
    float gq[32];
    {
        const float* gcp = gc + ((b * C + 8 * half) * HW + h) * HW + pix;
        #pragma unroll
        for (int ks = 0; ks < 2; ++ks)
            #pragma unroll
            for (int q = 0; q < 2; ++q)
                #pragma unroll
                for (int i = 0; i < 8; ++i)
                    gq[ks * 16 + q * 8 + i] = __builtin_fmaf(
                        gcp[(ks * 32 + q * 16 + i) * (HW * HW)], NL2E, -1.0f);
    }

    v8f acc[4] = {};

    #pragma unroll
    for (int ch = 0; ch < K9; ++ch) {            // chunk == 3x3 tap
        const int di = ch / 3 - 1;               // compile-time after unroll
        const int dj = ch % 3 - 1;
        const int hh = h + di;
        const int pw = pix + dj;
        const bool ok = ((unsigned)hh < (unsigned)HW) &&
                        ((unsigned)pw < (unsigned)HW);
        // clamped (always in-bounds) tap address; OOB lanes are zeroed via filter
        const int hhc = hh < 0 ? 0 : (hh > HW - 1 ? HW - 1 : hh);
        const int pwc = pw < 0 ? 0 : (pw > HW - 1 ? HW - 1 : pw);
        // OOB -> gsq = +inf -> exp2(inf)=inf -> rcp(inf+0.5)=0 -> act = x*0 = 0
        const float gsv = gs[((b * K9 + ch) * HW + h) * HW + pix];
        const float gsq = ok ? NL2E * gsv : __builtin_inff();
        const float* xp = x + ((b * C + 8 * half) * HW + hhc) * HW + pwc;

        // build both A fragments (16 pixels x 32 K each) directly in registers
        union { v16h v; v2h h2[8]; } A[2];
        #pragma unroll
        for (int ks = 0; ks < 2; ++ks)
            #pragma unroll
            for (int q = 0; q < 2; ++q)
                #pragma unroll
                for (int i2 = 0; i2 < 4; ++i2) {
                    int c0 = ks * 32 + q * 16 + i2 * 2;
                    int g0 = ks * 16 + q * 8 + i2 * 2;
                    // 1 + tanh(z) == rcp(exp2(-2log2e*z - 1) + 0.5)
                    float w0 = gq[g0] + gsq;
                    float w1 = gq[g0 + 1] + gsq;
                    float f0 = xp[c0 * (HW * HW)] *
                        __builtin_amdgcn_rcpf(__builtin_amdgcn_exp2f(w0) + 0.5f);
                    float f1 = xp[(c0 + 1) * (HW * HW)] *
                        __builtin_amdgcn_rcpf(__builtin_amdgcn_exp2f(w1) + 0.5f);
                    v2h pk = {(_Float16)f0, (_Float16)f1};
                    A[ks].h2[q * 4 + i2] = pk;
                }

        #pragma unroll
        for (int ks = 0; ks < 2; ++ks) {
            int s = ch * 2 + ks;
            #pragma unroll
            for (int og = 0; og < 4; ++og) {
                v16h bfr = *(const v16h*)
                    (wfrag + ((og * NSTEP + s) * 32 + lane) * 16);
                acc[og] = __builtin_amdgcn_wmma_f32_16x16x32_f16(
                    false, A[ks].v, false, bfr, (short)0, acc[og], false, false);
            }
        }
    }

    // ---- store transpose: D tile (pixel = i + 8*half, outch = og*16 + m) -> LDS ----
    const int p0 = wid * 16;
    #pragma unroll
    for (int og = 0; og < 4; ++og)
        #pragma unroll
        for (int i = 0; i < 8; ++i)
            lds_out[(og * 16 + m) * OSTR + p0 + i + 8 * half] = acc[og][i];
    __syncthreads();

    // coalesced writeback: lanes 0..31 cover 128 contiguous pixels per instruction
    const int pixc  = (t & 31) * 4;
    const int orow0 = (t >> 5) * 8;
    #pragma unroll
    for (int r = 0; r < 8; ++r) {
        int o = orow0 + r;
        v4f val = *(const v4f*)&lds_out[o * OSTR + pixc];
        *(v4f*)&out[((b * OC + o) * HW + h) * HW + pixc] = val;
    }
}

extern "C" void kernel_launch(void* const* d_in, const int* in_sizes, int n_in,
                              void* d_out, int out_size, void* d_ws, size_t ws_size,
                              hipStream_t stream) {
    (void)in_sizes; (void)n_in; (void)out_size;
    const float* x  = (const float*)d_in[0];
    const float* gc = (const float*)d_in[1];
    const float* gs = (const float*)d_in[2];
    const float* wt = (const float*)d_in[3];
    float* out = (float*)d_out;

    if (ws_size >= WFRAG_BYTES) {
        _Float16* wfg = (_Float16*)d_ws;
        paka_prep_kernel<<<dim3((NFRAG + 255) / 256), dim3(256), 0, stream>>>(wt, wfg);
        paka_wmma_kernel<true><<<dim3(8 * HW), dim3(256), 0, stream>>>(
            x, gc, gs, wt, wfg, out);
    } else {
        paka_wmma_kernel<false><<<dim3(8 * HW), dim3(256), 0, stream>>>(
            x, gc, gs, wt, nullptr, out);
    }
}
